// DeformConv3dPlusOffset_87660282511798
// MI455X (gfx1250) — matrix operand order repaired and checked by reading, not compiled, not run on hardware
//
#include <hip/hip_runtime.h>

// ---------------- problem constants ----------------
#define CIN   32
#define COUT  64
#define NTAP  27        // 3x3x3
#define VOX   32768     // 32^3
#define MPAD  96        // 81 offset channels padded to 6*16
#define PSTR  8         // padded stride for interp params (16B-aligned groups)

typedef __attribute__((ext_vector_type(16))) _Float16 v16h;
typedef __attribute__((ext_vector_type(8)))  _Float16 v8h;
typedef __attribute__((ext_vector_type(4)))  _Float16 v4h;
typedef __attribute__((ext_vector_type(8)))  float    v8f;

// Build v16h from two 8-half chunks (compiler folds into adjacent VGPRs)
__device__ __forceinline__ v16h pack16(const _Float16* lo, const _Float16* hi) {
    v8h a0 = *(const v8h*)lo;
    v8h a1 = *(const v8h*)hi;
    v16h r;
#pragma unroll
    for (int i = 0; i < 8; ++i) { r[i] = a0[i]; r[i + 8] = a1[i]; }
    return r;
}
// A-matrix 16x32 f16: per lane two chunks at +0B and +32B
__device__ __forceinline__ v16h load_a16(const _Float16* p) { return pack16(p, p + 16); }
// B-matrix 32x16 f16: per lane two chunks at +0B and +32B
__device__ __forceinline__ v16h load_b16(const _Float16* p) { return pack16(p, p + 16); }

// ---------------------------------------------------------------------------
// k_prep: f32 -> f16 weight repack.
//   wp_h[(n*96+o)*32+c] = (o<81) ? w_p[(o*32+c)*27+n] : 0
//   wd_h[(n*64+o)*32+c] = w_d[(o*32+c)*27+n]
// ---------------------------------------------------------------------------
__global__ void k_prep(const float* __restrict__ w_p, const float* __restrict__ w_d,
                       _Float16* __restrict__ wp_h, _Float16* __restrict__ wd_h) {
    const int NP = NTAP * MPAD * CIN;   // 82944
    const int ND = NTAP * COUT * CIN;   // 55296
    for (int i = blockIdx.x * blockDim.x + threadIdx.x; i < NP; i += gridDim.x * blockDim.x) {
        int c = i & 31;
        int o = (i >> 5) % MPAD;
        int n = i / (MPAD * CIN);
        float v = (o < 81) ? w_p[(o * CIN + c) * NTAP + n] : 0.0f;
        wp_h[i] = (_Float16)v;
    }
    for (int i = blockIdx.x * blockDim.x + threadIdx.x; i < ND; i += gridDim.x * blockDim.x) {
        int c = i & 31;
        int o = (i >> 5) % COUT;
        int n = i / (COUT * CIN);
        wd_h[i] = (_Float16)w_d[(o * CIN + c) * NTAP + n];
    }
}

// ---------------------------------------------------------------------------
// k_offset: offset conv as per-tap WMMA GEMM over a raw halo tile.
//   One block = 16 consecutive voxels (same z,y row half). 6 waves, each owns
//   16 of the 96 padded output rows. Taps address the halo tile directly.
// ---------------------------------------------------------------------------
__global__ void __launch_bounds__(192)
k_offset(const float* __restrict__ x, const _Float16* __restrict__ wp_h,
         const float* __restrict__ b_p, float* __restrict__ offs) {
    // halo tile t[zz:3][yy:3][xx:18][c:32] f16 (xx covers x0-1 .. x0+16)
    __shared__ __align__(16) _Float16 t[3 * 3 * 18 * CIN];   // 10368 B

    const int tid = threadIdx.x;
    const int v0  = blockIdx.x * 16;
    const int z   = v0 >> 10;
    const int y   = (v0 >> 5) & 31;
    const int x0  = v0 & 31;

    // stage halo tile, xx fastest across lanes for coalesced reads
    for (int idx = tid; idx < 3 * 3 * 18 * CIN; idx += 192) {
        int xx = idx % 18;
        int r  = idx / 18;
        int c  = r & 31;
        int yy = (r >> 5) % 3;
        int zz = (r >> 5) / 3;
        int gz = z + zz - 1, gy = y + yy - 1, gx = x0 + xx - 1;
        float val = 0.0f;
        if (((unsigned)gz < 32u) & ((unsigned)gy < 32u) & ((unsigned)gx < 32u))
            val = x[((c * 32 + gz) * 32 + gy) * 32 + gx];
        t[((zz * 3 + yy) * 18 + xx) * CIN + c] = (_Float16)val;
    }
    __syncthreads();

    const int lane = tid & 31;
    const int wid  = tid >> 5;            // 0..5
    const int nn   = lane & 15;           // N index (voxel)
    const int hi   = lane >> 4;           // half-wave
    const int orow = 16 * wid + nn;       // A-matrix M row

    // fully unrolled 27-tap WMMA chain: all addresses are base + const offset
    const _Float16* aB = wp_h + (size_t)orow * CIN + hi * 8;
    const _Float16* bB = t + nn * CIN + hi * 8;
    v8f c = {};
#pragma unroll
    for (int n = 0; n < NTAP; ++n) {
        const int kd = n / 9, kh = (n / 3) % 3, kw = n % 3;   // constants after unroll
        v16h a = load_a16(aB + (size_t)n * MPAD * CIN);
        v16h b = load_b16(bB + ((kd * 3 + kh) * 18 + kw) * CIN);
        c = __builtin_amdgcn_wmma_f32_16x16x32_f16(false, a, false, b,
                                                   (short)0, c, false, false);
    }

    const int hi8 = hi * 8;
    if (wid < 5) {                                  // wave-uniform: all rows valid
        float* op = offs + (size_t)(16 * wid + hi8) * VOX + v0 + nn;
#pragma unroll
        for (int r = 0; r < 8; ++r)
            op[(size_t)r * VOX] = c[r] + b_p[16 * wid + hi8 + r];
    } else if (hi == 0) {                           // wave 5: only o==80 valid
        offs[(size_t)80 * VOX + v0 + nn] = c[0] + b_p[80];
    }
}

// ---------------------------------------------------------------------------
// k_deform: fused trilinear sampling + output GEMM.
//   One block = 16 voxels; 4 waves, each owns 16 of 64 output channels.
// ---------------------------------------------------------------------------
__global__ void __launch_bounds__(128)
k_deform(const float* __restrict__ x, const float* __restrict__ offs,
         const _Float16* __restrict__ wd_h, float* __restrict__ out) {
    __shared__ __align__(16) _Float16 s2[NTAP * 16 * CIN];   // [n][v][c] f16 samples
    __shared__ __align__(16) int   ip[NTAP * 16 * PSTR];     // q0d,q1d,q0h,q1h,q0w,q1w,pad
    __shared__ __align__(16) float fp[NTAP * 16 * PSTR];     // w0d,w1d,w0h,w1h,w0w,w1w,pad

    const int tid = threadIdx.x;
    const int v0  = blockIdx.x * 16;
    const int z   = v0 >> 10;
    const int y   = (v0 >> 5) & 31;
    const int x0  = v0 & 31;

    // Phase 1: interpolation parameters per (tap, voxel) — replicate reference math
    for (int item = tid; item < NTAP * 16; item += 128) {
        int n = item >> 4;
        int v = item & 15;
        int vox = v0 + v;
        float od = offs[(size_t)(0 * NTAP + n) * VOX + vox];
        float oh = offs[(size_t)(1 * NTAP + n) * VOX + vox];
        float ow = offs[(size_t)(2 * NTAP + n) * VOX + vox];
        int dz = n / 9 - 1, dh = (n / 3) % 3 - 1, dw = n % 3 - 1;
        float pd = (float)(z + 1 + dz) + od;
        float ph = (float)(y + 1 + dh) + oh;
        float pw = (float)(x0 + v + 1 + dw) + ow;
        pd = fminf(fmaxf(pd, 0.0f), 33.0f);
        ph = fminf(fmaxf(ph, 0.0f), 33.0f);
        pw = fminf(fmaxf(pw, 0.0f), 33.0f);
        int q0d = min(max((int)floorf(pd), 0), 33), q1d = min(q0d + 1, 33);
        int q0h = min(max((int)floorf(ph), 0), 33), q1h = min(q0h + 1, 33);
        int q0w = min(max((int)floorf(pw), 0), 33), q1w = min(q0w + 1, 33);
        int*   qi = ip + item * PSTR;
        float* fw = fp + item * PSTR;
        qi[0] = q0d; qi[1] = q1d; qi[2] = q0h; qi[3] = q1h; qi[4] = q0w; qi[5] = q1w;
        fw[0] = 1.0f + ((float)q0d - pd);  fw[1] = 1.0f - ((float)q1d - pd);
        fw[2] = 1.0f + ((float)q0h - ph);  fw[3] = 1.0f - ((float)q1h - ph);
        fw[4] = 1.0f + ((float)q0w - pw);  fw[5] = 1.0f - ((float)q1w - pw);
    }
    __syncthreads();

    // Phase 2: 8-corner gather-lerp, 4 channels per item (shared corner params),
    // voxel fastest across lanes so gathers are near-contiguous.
    for (int idx = tid; idx < NTAP * 8 * 16; idx += 128) {
        int n  = idx >> 7;
        int c4 = (idx >> 4) & 7;          // channel group: c = 4*c4 .. +3
        int v  = idx & 15;
        const int*   qi = ip + (n * 16 + v) * PSTR;
        const float* fw = fp + (n * 16 + v) * PSTR;
        const float* xc = x + (size_t)(c4 * 4) * VOX;
        float acc0 = 0.f, acc1 = 0.f, acc2 = 0.f, acc3 = 0.f;
#pragma unroll
        for (int sd = 0; sd < 2; ++sd)
#pragma unroll
            for (int sh = 0; sh < 2; ++sh)
#pragma unroll
                for (int sw = 0; sw < 2; ++sw) {
                    int zz = qi[sd] - 1, yy = qi[2 + sh] - 1, ww = qi[4 + sw] - 1;
                    float wt = fw[sd] * fw[2 + sh] * fw[4 + sw];
                    if (((unsigned)zz < 32u) & ((unsigned)yy < 32u) & ((unsigned)ww < 32u)) {
                        int base = (zz * 32 + yy) * 32 + ww;
                        acc0 += wt * xc[base];
                        acc1 += wt * xc[base + VOX];
                        acc2 += wt * xc[base + 2 * VOX];
                        acc3 += wt * xc[base + 3 * VOX];
                    }
                }
        v4h pk4;   // one 64-bit DS store for 4 channels
        pk4[0] = (_Float16)acc0; pk4[1] = (_Float16)acc1;
        pk4[2] = (_Float16)acc2; pk4[3] = (_Float16)acc3;
        *(v4h*)(s2 + n * 512 + v * 32 + c4 * 4) = pk4;
    }
    __syncthreads();

    // Phase 3: fully unrolled 27-WMMA chain per wave
    const int lane = tid & 31;
    const int wid  = tid >> 5;           // 0..3
    const int nn   = lane & 15;
    const int hi   = lane >> 4;
    const int orow = 16 * wid + nn;

    const _Float16* aB = wd_h + (size_t)orow * CIN + hi * 8;
    const _Float16* bB = s2 + nn * 32 + hi * 8;
    v8f c = {};
#pragma unroll
    for (int n = 0; n < NTAP; ++n) {
        v16h a = load_a16(aB + (size_t)n * COUT * CIN);
        v16h b = load_b16(bB + n * 512);
        c = __builtin_amdgcn_wmma_f32_16x16x32_f16(false, a, false, b,
                                                   (short)0, c, false, false);
    }

    const int hi8 = hi * 8;
    float* op = out + (size_t)(16 * wid + hi8) * VOX + v0 + nn;
#pragma unroll
    for (int r = 0; r < 8; ++r)
        op[(size_t)r * VOX] = c[r];
}

// ---------------------------------------------------------------------------
extern "C" void kernel_launch(void* const* d_in, const int* in_sizes, int n_in,
                              void* d_out, int out_size, void* d_ws, size_t ws_size,
                              hipStream_t stream) {
    const float* x   = (const float*)d_in[0];   // (1,32,32,32,32)
    const float* w_p = (const float*)d_in[1];   // (81,32,3,3,3)
    const float* b_p = (const float*)d_in[2];   // (81,)
    const float* w_d = (const float*)d_in[3];   // (64,32,3,3,3)
    float* out = (float*)d_out;                 // (1,64,32,32,32)

    char* ws = (char*)d_ws;
    float*    offs = (float*)ws;                                  // 81*32768 f32
    _Float16* wp_h = (_Float16*)(ws + (size_t)81 * VOX * 4);      // 27*96*32 f16
    _Float16* wd_h = wp_h + NTAP * MPAD * CIN;                    // 27*64*32 f16

    k_prep  <<<96,   256, 0, stream>>>(w_p, w_d, wp_h, wd_h);
    k_offset<<<VOX / 16, 192, 0, stream>>>(x, wp_h, b_p, offs);
    k_deform<<<VOX / 16, 128, 0, stream>>>(x, offs, wd_h, out);
}
